// GATMod_26044681682946
// MI455X (gfx1250) — matrix operand order repaired and checked by reading, not compiled, not run on hardware
//
#include <hip/hip_runtime.h>
#include <hip/hip_bf16.h>

typedef __bf16 bf16_t;
typedef __attribute__((ext_vector_type(16))) __bf16 v16bf;
typedef __attribute__((ext_vector_type(8)))  float  v8f;

#define N_NODES 20000
#define N_EDGES 320000
#define DIM_IN  256
#define DIM_HID 64
#define N_HEAD  4
#define DIM_C   64
#define DIM_HC  256
#define DIM_FF  128
#define N_LAYER 4

// ---------------------------------------------------------------------------
// helpers
// ---------------------------------------------------------------------------
__device__ inline void atomic_add_f32(float* p, float v) {
    // hardware global f32 atomic add (no CAS loop)
    asm volatile("global_atomic_add_f32 %0, %1, off" :: "v"(p), "v"(v) : "memory");
}

__device__ inline unsigned enc_f32(float f) {   // order-preserving uint encoding
    unsigned u = __float_as_uint(f);
    return (u & 0x80000000u) ? ~u : (u | 0x80000000u);
}
__device__ inline float dec_f32(unsigned k) {
    unsigned u = (k & 0x80000000u) ? (k & 0x7FFFFFFFu) : ~k;
    return __uint_as_float(u);
}

// ---------------------------------------------------------------------------
// f32 -> bf16 weight conversion
// ---------------------------------------------------------------------------
__global__ void cvt_bf16_kernel(const float* __restrict__ src, bf16_t* __restrict__ dst, int n) {
    int i = blockIdx.x * blockDim.x + threadIdx.x;
    if (i < n) dst[i] = (bf16_t)src[i];
}

// ---------------------------------------------------------------------------
// WMMA GEMM:  D[M,N] = act(A[M,K](f32->bf16) @ B[K,N](bf16) + bias) * scale
// block = 128 threads (4 waves). wave w -> rows [blk*64 + w*16, +16),
// cols [blockIdx.y*64, +64). K multiple of 32 (<=256), N multiple of 64.
// B is staged into LDS pre-swizzled into WMMA fragment layout so each lane's
// 16-element fragment is one 32-byte contiguous LDS read (2x ds_load_b128).
// act: 0 = none, 1 = exact GELU
// ---------------------------------------------------------------------------
__global__ __launch_bounds__(128)
void gemm_bf16_wmma_kernel(const float* __restrict__ A, const bf16_t* __restrict__ B,
                           const float* __restrict__ bias, float* __restrict__ D,
                           int M, int K, int N, float scale, int act)
{
    // fragment layout: [kTile(<=8)][nTile(4)][lane(32)][elem(16)] bf16 = 32KB max
    __shared__ alignas(32) bf16_t Bs[(256 / 32) * 4 * 32 * 16];

    const int tid     = threadIdx.x;
    const int lane    = tid & 31;
    const int wave    = tid >> 5;
    const int colBase = blockIdx.y * 64;
    const int rowBase = blockIdx.x * 64 + wave * 16;

    // cooperative stage of B[0:K, colBase:colBase+64] into fragment layout.
    // ISA B layout (32x16 bf16): lanes 0-15 hold K 0..7,16..23, lanes 16-31 hold
    // K 8..15,24..31; lane&15 selects column; element j = (k&7)|((k>>4)<<3).
    for (int idx = tid; idx < K * 64; idx += 128) {
        int k  = idx >> 6;                     // global-k (coalesced over n)
        int n  = idx & 63;
        int kt = k >> 5;
        int kr = k & 31;
        int t  = n >> 4;
        int li = (((kr >> 3) & 1) << 4) | (n & 15);  // fragment lane
        int j  = (kr & 7) | ((kr >> 4) << 3);        // fragment element
        Bs[((((kt << 2) | t) << 5) + li) * 16 + j] = B[k * N + colBase + n];
    }
    __syncthreads();

    const int  half = lane >> 4;
    const int  l15  = lane & 15;
    const int  m    = rowBase + l15;           // A row owned by this lane
    const bool mvalid = (m < M);

    if (mvalid) __builtin_prefetch(&A[(size_t)m * K], 0, 0);

    const float4* Arow = reinterpret_cast<const float4*>(A + (size_t)m * K);
    const v16bf*  Bf   = reinterpret_cast<const v16bf*>(Bs);

    v8f acc[4] = {};

    for (int kk = 0; kk < K; kk += 32) {
        // ---- A fragment (16x32 bf16): two aligned 32B runs per lane ----
        // elements 0..7  -> K = kk + half*8 .. +7
        // elements 8..15 -> K = kk + 16 + half*8 .. +7
        float4 z4 = make_float4(0.f, 0.f, 0.f, 0.f);
        int b0 = (kk + half * 8) >> 2;
        int b1 = (kk + 16 + half * 8) >> 2;
        float4 p0 = mvalid ? Arow[b0]     : z4;
        float4 p1 = mvalid ? Arow[b0 + 1] : z4;
        float4 p2 = mvalid ? Arow[b1]     : z4;
        float4 p3 = mvalid ? Arow[b1 + 1] : z4;
        v16bf a;
        a[0]  = (bf16_t)p0.x; a[1]  = (bf16_t)p0.y; a[2]  = (bf16_t)p0.z; a[3]  = (bf16_t)p0.w;
        a[4]  = (bf16_t)p1.x; a[5]  = (bf16_t)p1.y; a[6]  = (bf16_t)p1.z; a[7]  = (bf16_t)p1.w;
        a[8]  = (bf16_t)p2.x; a[9]  = (bf16_t)p2.y; a[10] = (bf16_t)p2.z; a[11] = (bf16_t)p2.w;
        a[12] = (bf16_t)p3.x; a[13] = (bf16_t)p3.y; a[14] = (bf16_t)p3.z; a[15] = (bf16_t)p3.w;

        // ---- 4 B fragments: one 32B LDS vector read each + WMMA ----
        const int ktBase = (kk >> 5) << 2;
        #pragma unroll
        for (int t = 0; t < 4; ++t) {
            v16bf b = Bf[((ktBase + t) << 5) + lane];
            acc[t] = __builtin_amdgcn_wmma_f32_16x16x32_bf16(
                         false, a, false, b, (short)0, acc[t], false, false);
        }
    }

    // ---- epilogue: bias, activation, scale, store (D layout: m = r + 8*half) ----
    #pragma unroll
    for (int t = 0; t < 4; ++t) {
        int n  = colBase + t * 16 + l15;
        float bv = bias ? bias[n] : 0.0f;
        #pragma unroll
        for (int r = 0; r < 8; ++r) {
            int mm = rowBase + r + 8 * half;
            if (mm < M) {
                float v = acc[t][r] + bv;
                if (act == 1) v = 0.5f * v * (1.0f + erff(v * 0.70710678118f));
                D[(size_t)mm * N + n] = v * scale;
            }
        }
    }
}

// ---------------------------------------------------------------------------
// attention scores: a_s[n,h] = xh[n,h,:].att_src ; a_d likewise (wave per (n,h))
// ---------------------------------------------------------------------------
__global__ __launch_bounds__(256)
void attn_score_kernel(const float* __restrict__ xh, const float* __restrict__ att_s,
                       const float* __restrict__ att_d, float* __restrict__ a_s,
                       float* __restrict__ a_d, int n_nodes)
{
    int gw   = blockIdx.x * 8 + (threadIdx.x >> 5);
    int lane = threadIdx.x & 31;
    int node = gw >> 2;
    int head = gw & 3;
    if (node >= n_nodes) return;

    float s = 0.f, d = 0.f;
    #pragma unroll
    for (int k = 0; k < 2; ++k) {
        int c  = k * 32 + lane;
        float v = xh[(size_t)node * DIM_HC + head * DIM_C + c];
        s += v * att_s[head * DIM_C + c];
        d += v * att_d[head * DIM_C + c];
    }
    #pragma unroll
    for (int off = 16; off > 0; off >>= 1) {
        s += __shfl_xor(s, off, 32);
        d += __shfl_xor(d, off, 32);
    }
    if (lane == 0) {
        a_s[node * 4 + head] = s;
        a_d[node * 4 + head] = d;
    }
}

// ---------------------------------------------------------------------------
// per-layer init: segment-max buffer (encoded 0 < any real score), z = 0, out = bc
// ---------------------------------------------------------------------------
__global__ void init_layer_kernel(unsigned* __restrict__ m_enc, float* __restrict__ z,
                                  float* __restrict__ out, const float* __restrict__ bc,
                                  int n_nodes)
{
    int i = blockIdx.x * blockDim.x + threadIdx.x;
    if (i < n_nodes * N_HEAD) { m_enc[i] = 0u; z[i] = 0.f; }
    if (i < n_nodes * DIM_HC) out[i] = bc[i & (DIM_HC - 1)];
}

// ---------------------------------------------------------------------------
// edge pass 1: e = leaky_relu(a_s[src]+a_d[dst]); store e; segment max via atomicMax
// ---------------------------------------------------------------------------
__global__ void edge_max_kernel(const int* __restrict__ ei, const float* __restrict__ a_s,
                                const float* __restrict__ a_d, float* __restrict__ ebuf,
                                unsigned* __restrict__ m_enc, int E, int T)
{
    int i = blockIdx.x * blockDim.x + threadIdx.x;
    if (i >= T) return;
    int s = (i < E) ? ei[i]     : (i - E);   // self loops appended
    int d = (i < E) ? ei[E + i] : (i - E);
    #pragma unroll
    for (int h = 0; h < N_HEAD; ++h) {
        float e = a_s[s * 4 + h] + a_d[d * 4 + h];
        e = (e > 0.f) ? e : 0.2f * e;
        ebuf[(size_t)i * 4 + h] = e;
        atomicMax(&m_enc[d * 4 + h], enc_f32(e));
    }
}

// ---------------------------------------------------------------------------
// edge pass 2: e = exp(e - max[dst]); segment sum via hw f32 atomics
// ---------------------------------------------------------------------------
__global__ void edge_expsum_kernel(const int* __restrict__ ei, float* __restrict__ ebuf,
                                   const unsigned* __restrict__ m_enc, float* __restrict__ z,
                                   int E, int T)
{
    int i = blockIdx.x * blockDim.x + threadIdx.x;
    if (i >= T) return;
    int d = (i < E) ? ei[E + i] : (i - E);
    #pragma unroll
    for (int h = 0; h < N_HEAD; ++h) {
        float e  = ebuf[(size_t)i * 4 + h];
        float mx = dec_f32(m_enc[d * 4 + h]);
        float ex = __expf(e - mx);
        ebuf[(size_t)i * 4 + h] = ex;
        atomic_add_f32(&z[d * 4 + h], ex);
    }
}

// ---------------------------------------------------------------------------
// edge pass 3: out[dst,:] += xh[src,:] * alpha   (wave per edge, 8 edges/block)
// ---------------------------------------------------------------------------
__global__ __launch_bounds__(256)
void edge_scatter_kernel(const int* __restrict__ ei, const float* __restrict__ xh,
                         const float* __restrict__ ebuf, const float* __restrict__ z,
                         float* __restrict__ out, int E, int T)
{
    int i    = blockIdx.x * 8 + (threadIdx.x >> 5);
    int lane = threadIdx.x & 31;
    if (i >= T) return;
    int s = (i < E) ? ei[i]     : (i - E);
    int d = (i < E) ? ei[E + i] : (i - E);
    #pragma unroll
    for (int h = 0; h < N_HEAD; ++h) {
        float alpha = ebuf[(size_t)i * 4 + h] / (z[d * 4 + h] + 1e-16f);
        #pragma unroll
        for (int k = 0; k < 2; ++k) {
            int c = h * DIM_C + k * 32 + lane;
            atomic_add_f32(&out[(size_t)d * DIM_HC + c],
                           xh[(size_t)s * DIM_HC + c] * alpha);
        }
    }
}

// ---------------------------------------------------------------------------
// h += LayerNorm(f2) * g + b   (thread per row)
// ---------------------------------------------------------------------------
__global__ void ln_residual_kernel(const float* __restrict__ f2, const float* __restrict__ g,
                                   const float* __restrict__ b, float* __restrict__ h,
                                   int n_nodes)
{
    int row = blockIdx.x * blockDim.x + threadIdx.x;
    if (row >= n_nodes) return;
    const float* r = f2 + (size_t)row * DIM_HID;
    float mean = 0.f;
    for (int j = 0; j < DIM_HID; ++j) mean += r[j];
    mean *= (1.0f / DIM_HID);
    float var = 0.f;
    for (int j = 0; j < DIM_HID; ++j) { float dd = r[j] - mean; var += dd * dd; }
    var *= (1.0f / DIM_HID);
    float inv = rsqrtf(var + 1e-5f);
    for (int j = 0; j < DIM_HID; ++j)
        h[(size_t)row * DIM_HID + j] += (r[j] - mean) * inv * g[j] + b[j];
}

// ---------------------------------------------------------------------------
// launch
// ---------------------------------------------------------------------------
extern "C" void kernel_launch(void* const* d_in, const int* in_sizes, int n_in,
                              void* d_out, int out_size, void* d_ws, size_t ws_size,
                              hipStream_t stream)
{
    const float* x       = (const float*)d_in[0];
    const int*   ei      = (const int*)  d_in[1];   // [2,E]
    const float* We      = (const float*)d_in[2];
    const float* be      = (const float*)d_in[3];
    const float* Wc      = (const float*)d_in[4];   // [L,64,256]
    const float* att_src = (const float*)d_in[5];   // [L,4,64]
    const float* att_dst = (const float*)d_in[6];
    const float* bc      = (const float*)d_in[7];   // [L,256]
    const float* W1      = (const float*)d_in[8];   // [L,256,128]
    const float* b1      = (const float*)d_in[9];
    const float* W2      = (const float*)d_in[10];  // [L,128,64]
    const float* b2      = (const float*)d_in[11];
    const float* ln_g    = (const float*)d_in[12];
    const float* ln_b    = (const float*)d_in[13];

    float* h = (float*)d_out;                        // [N,64] accumulated in place

    const int Nn = N_NODES, E = N_EDGES, T = E + Nn;

    // ---- workspace carving ----
    char*  ws  = (char*)d_ws;
    size_t off = 0;
    auto carve = [&](size_t bytes) -> void* {
        void* p = ws + off;
        off += (bytes + 255) & ~(size_t)255;
        return p;
    };
    bf16_t*   We_b  = (bf16_t*)  carve(sizeof(bf16_t) * DIM_IN * DIM_HID);
    bf16_t*   Wc_b  = (bf16_t*)  carve(sizeof(bf16_t) * N_LAYER * DIM_HID * DIM_HC);
    bf16_t*   W1_b  = (bf16_t*)  carve(sizeof(bf16_t) * N_LAYER * DIM_HC * DIM_FF);
    bf16_t*   W2_b  = (bf16_t*)  carve(sizeof(bf16_t) * N_LAYER * DIM_FF * DIM_HID);
    float*    xh    = (float*)   carve(sizeof(float) * (size_t)Nn * DIM_HC);
    float*    a_s   = (float*)   carve(sizeof(float) * (size_t)Nn * N_HEAD);
    float*    a_d   = (float*)   carve(sizeof(float) * (size_t)Nn * N_HEAD);
    unsigned* m_enc = (unsigned*)carve(sizeof(unsigned) * (size_t)Nn * N_HEAD);
    float*    z     = (float*)   carve(sizeof(float) * (size_t)Nn * N_HEAD);
    float*    ebuf  = (float*)   carve(sizeof(float) * (size_t)T * N_HEAD);
    float*    outb  = (float*)   carve(sizeof(float) * (size_t)Nn * DIM_HC);
    float*    f1    = (float*)   carve(sizeof(float) * (size_t)Nn * DIM_FF);
    float*    f2    = (float*)   carve(sizeof(float) * (size_t)Nn * DIM_HID);

    // ---- weights to bf16 ----
    auto cvt = [&](const float* s, bf16_t* dst, int n) {
        cvt_bf16_kernel<<<(n + 255) / 256, 256, 0, stream>>>(s, dst, n);
    };
    cvt(We, We_b, DIM_IN * DIM_HID);
    cvt(Wc, Wc_b, N_LAYER * DIM_HID * DIM_HC);
    cvt(W1, W1_b, N_LAYER * DIM_HC * DIM_FF);
    cvt(W2, W2_b, N_LAYER * DIM_FF * DIM_HID);

    const int rowBlocks = (Nn + 63) / 64;

    // ---- embed: h = (x @ We + be) * sqrt(64) ----
    gemm_bf16_wmma_kernel<<<dim3(rowBlocks, DIM_HID / 64), 128, 0, stream>>>(
        x, We_b, be, h, Nn, DIM_IN, DIM_HID, 8.0f, 0);

    for (int l = 0; l < N_LAYER; ++l) {
        // xh = h @ Wc[l]
        gemm_bf16_wmma_kernel<<<dim3(rowBlocks, DIM_HC / 64), 128, 0, stream>>>(
            h, Wc_b + (size_t)l * DIM_HID * DIM_HC, nullptr, xh,
            Nn, DIM_HID, DIM_HC, 1.0f, 0);

        // attention logits per node/head
        attn_score_kernel<<<(Nn * N_HEAD + 7) / 8, 256, 0, stream>>>(
            xh, att_src + l * N_HEAD * DIM_C, att_dst + l * N_HEAD * DIM_C,
            a_s, a_d, Nn);

        // reset segment buffers; out = bc[l]
        init_layer_kernel<<<(Nn * DIM_HC + 255) / 256, 256, 0, stream>>>(
            m_enc, z, outb, bc + l * DIM_HC, Nn);

        // segment softmax over incoming edges
        edge_max_kernel<<<(T + 255) / 256, 256, 0, stream>>>(ei, a_s, a_d, ebuf, m_enc, E, T);
        edge_expsum_kernel<<<(T + 255) / 256, 256, 0, stream>>>(ei, ebuf, m_enc, z, E, T);
        edge_scatter_kernel<<<(T + 7) / 8, 256, 0, stream>>>(ei, xh, ebuf, z, outb, E, T);

        // FFN
        gemm_bf16_wmma_kernel<<<dim3(rowBlocks, DIM_FF / 64), 128, 0, stream>>>(
            outb, W1_b + (size_t)l * DIM_HC * DIM_FF, b1 + l * DIM_FF, f1,
            Nn, DIM_HC, DIM_FF, 1.0f, 1 /*gelu*/);
        gemm_bf16_wmma_kernel<<<dim3(rowBlocks, DIM_HID / 64), 128, 0, stream>>>(
            f1, W2_b + (size_t)l * DIM_FF * DIM_HID, b2 + l * DIM_HID, f2,
            Nn, DIM_FF, DIM_HID, 1.0f, 0);

        // residual + layernorm
        ln_residual_kernel<<<(Nn + 255) / 256, 256, 0, stream>>>(
            f2, ln_g + l * DIM_HID, ln_b + l * DIM_HID, h, Nn);
    }
}
